// DygLearner_68788196213096
// MI455X (gfx1250) — matrix operand order repaired and checked by reading, not compiled, not run on hardware
//
#include <hip/hip_runtime.h>
#include <hip/hip_bf16.h>

// ---------------------------------------------------------------------------
// DygLearner forward for MI455X (gfx1250, wave32, WMMA).
// All matmul-shaped work runs through v_wmma_f32_16x16x32_f16.
// ---------------------------------------------------------------------------

typedef __attribute__((ext_vector_type(16))) _Float16 v16h;
typedef __attribute__((ext_vector_type(8)))  _Float16 v8h;
typedef __attribute__((ext_vector_type(8)))  float    v8f;

#define B_   8
#define N_   400
#define T_   16
#define L_   64
#define C_   96
#define E_   64
#define NE_  200
#define BT_  128          // B*T
#define MROWS_ 51200      // B*N*T
#define SAMP_  3200       // B*N

// ---- WMMA fragment loaders (16-bit, wave32) --------------------------------
// A 16x32 (MxK): lane m=l&15, h=l>>4 holds K chunk [16h,16h+16)
// B 32x16 (KxN) stored as Bt[N][K] row-major: lane n=l&15 holds K chunk [16h,16h+16)
__device__ __forceinline__ v16h load_a(const _Float16* tile, int ld, int k0) {
    int lane = threadIdx.x & 31;
    int m = lane & 15, h = lane >> 4;
    const _Float16* p = tile + (long)m * ld + k0;
    v8h c0 = *(const v8h*)(p + 16 * h);
    v8h c1 = *(const v8h*)(p + 16 * h + 8);
    v16h r;
#pragma unroll
    for (int i = 0; i < 8; ++i) { r[i] = c0[i]; r[8 + i] = c1[i]; }
    return r;
}
__device__ __forceinline__ v16h load_b(const _Float16* tile, int ld, int k0) {
    int lane = threadIdx.x & 31;
    int n = lane & 15, h = lane >> 4;
    return *(const v16h*)(tile + (long)n * ld + k0 + 16 * h);
}
__device__ __forceinline__ v8f wmma32(v16h a, v16h b, v8f c) {
    return __builtin_amdgcn_wmma_f32_16x16x32_f16(false, a, false, b,
                                                  (short)0, c, false, false);
}

// ---- f32 -> f16 convert ----------------------------------------------------
__global__ void k_cvt(const float* __restrict__ src, _Float16* __restrict__ dst, int n) {
    for (int i = blockIdx.x * blockDim.x + threadIdx.x; i < n;
         i += gridDim.x * blockDim.x)
        dst[i] = (_Float16)src[i];
}

// ---- conv weight re-layout: src[o][c][k] f32 -> dst[k][o][c] f16 -----------
__global__ void k_convw(const float* __restrict__ src, _Float16* __restrict__ dst, int Kg) {
    int tot = 32 * 96 * Kg;
    for (int t = blockIdx.x * blockDim.x + threadIdx.x; t < tot;
         t += gridDim.x * blockDim.x) {
        int o = t / (96 * Kg);
        int r = t - o * 96 * Kg;
        int c = r / Kg;
        int k = r - c * Kg;
        dst[k * 3072 + o * 96 + c] = (_Float16)src[t];
    }
}

// ---- generic WMMA GEMM: OUT = act(A[M,K] * Bt[Nout,K]^T + bias) ------------
// mode 0: f16 row-major [M][Nout]   mode 1: f32 scatter into [B,T,N,E]
__global__ void __launch_bounds__(256)
k_gemm(const _Float16* __restrict__ A, const _Float16* __restrict__ Bt,
       const float* __restrict__ bias, _Float16* __restrict__ outh,
       float* __restrict__ outf, int K, int Nout, int relu, int mode) {
    int lane = threadIdx.x & 31;
    int wave = threadIdx.x >> 5;
    int tiles_n = Nout >> 4;
    long tile = (long)blockIdx.x * 8 + wave;
    long ntiles = (long)(MROWS_ / 16) * tiles_n;
    if (tile >= ntiles) return;                 // wave-uniform
    int rt = (int)(tile / tiles_n);
    int ct = (int)(tile % tiles_n);
    const _Float16* At = A + (long)rt * 16 * K;
    const _Float16* Bti = Bt + (long)ct * 16 * K;
    v8f acc = {};
    for (int k0 = 0; k0 < K; k0 += 32)
        acc = wmma32(load_a(At, K, k0), load_b(Bti, K, k0), acc);
    int n = lane & 15, h = lane >> 4;
    float bn = bias ? bias[ct * 16 + n] : 0.f;
#pragma unroll
    for (int i = 0; i < 8; ++i) {
        int r = rt * 16 + i + 8 * h;
        float v = acc[i] + bn;
        if (relu) v = fmaxf(v, 0.f);
        if (mode == 0) {
            outh[(long)r * Nout + ct * 16 + n] = (_Float16)v;
        } else {
            int b = r / (N_ * T_);
            int nn = (r / T_) % N_;
            int t = r % T_;
            outf[(((long)b * T_ + t) * N_ + nn) * E_ + ct * 16 + n] = v;
        }
    }
}

// ---- causal dilated group conv via WMMA ------------------------------------
// X,Y: f16 [3200][16][96]; W: 11 tap blocks of [32][96] f16 (layer base)
__global__ void __launch_bounds__(256)
k_conv(const _Float16* __restrict__ X, _Float16* __restrict__ Y,
       const _Float16* __restrict__ W, const float* __restrict__ b0,
       const float* __restrict__ b1, const float* __restrict__ b2, int dil) {
    __shared__ _Float16 lds[8][26 * 96];        // per-wave padded sample
    int lane = threadIdx.x & 31, wave = threadIdx.x >> 5;
    int s = blockIdx.x * 8 + wave;
    int pad = 5 * dil;
    _Float16* xp = lds[wave];
    int nch = (pad + 16) * 12;                  // 8-half chunks
    for (int ci = lane; ci < nch; ci += 32) {
        int row = ci / 12, cc = (ci % 12) * 8;
        v8h v;
        if (row < pad) {
#pragma unroll
            for (int q = 0; q < 8; ++q) v[q] = (_Float16)0.f;
        } else {
            v = *(const v8h*)(X + ((long)s * 16 + row - pad) * 96 + cc);
        }
        *(v8h*)(xp + row * 96 + cc) = v;
    }
    __syncthreads();
    const int taps[3]   = {2, 3, 6};
    const int tapoff[3] = {0, 2, 5};
    const float* bptr[3] = {b0, b1, b2};
#pragma unroll
    for (int g = 0; g < 3; ++g) {
#pragma unroll
        for (int mt = 0; mt < 2; ++mt) {
            v8f acc = {};
            for (int k = 0; k < taps[g]; ++k) {
                const _Float16* wt = W + (tapoff[g] + k) * 3072 + mt * 16 * 96;
                // correlation: tap k reads x[c, t - (K-1-k)*dil]
                const _Float16* xb = xp + (pad - (taps[g] - 1 - k) * dil) * 96;
#pragma unroll
                for (int kc = 0; kc < 96; kc += 32)
                    acc = wmma32(load_a(wt, 96, kc), load_b(xb, 96, kc), acc);
            }
            int t = lane & 15, h = lane >> 4;
            const float* bb = bptr[g];
            v8h o;
#pragma unroll
            for (int i = 0; i < 8; ++i) {
                float v = acc[i] + bb[mt * 16 + i + 8 * h];
                o[i] = (_Float16)fmaxf(v, 0.f);
            }
            *(v8h*)(Y + ((long)s * 16 + t) * 96 + g * 32 + mt * 16 + 8 * h) = o;
        }
    }
}

// ---- spatial attention + x_att (f16) ---------------------------------------
__global__ void __launch_bounds__(256)
k_attn(const float* __restrict__ nf, const float* __restrict__ w1,
       const float* __restrict__ w2, _Float16* __restrict__ xatt) {
    __shared__ float m[N_];
    __shared__ float a1[NE_];
    __shared__ float av[N_];
    int bt = blockIdx.x;
    const float* xb = nf + (long)bt * N_ * E_;
    for (int n = threadIdx.x; n < N_; n += 256) {
        float s = 0.f;
        for (int e = 0; e < E_; ++e) s += xb[n * E_ + e];
        m[n] = s * (1.f / (float)E_);
    }
    __syncthreads();
    for (int mm = threadIdx.x; mm < NE_; mm += 256) {
        float s = 0.f;
        const float* wr = w1 + (long)mm * N_;
        for (int n = 0; n < N_; ++n) s += m[n] * wr[n];
        a1[mm] = fmaxf(s, 0.f);
    }
    __syncthreads();
    for (int n = threadIdx.x; n < N_; n += 256) {
        float s = 0.f;
        const float* wr = w2 + (long)n * NE_;
        for (int k = 0; k < NE_; ++k) s += a1[k] * wr[k];
        av[n] = 1.f / (1.f + __expf(-s));
    }
    __syncthreads();
    for (int idx = threadIdx.x; idx < N_ * E_; idx += 256) {
        int n = idx >> 6;
        xatt[(long)bt * N_ * E_ + idx] = (_Float16)(av[n] * xb[idx]);
    }
}

// ---- scores = relu(Xa Xa^T), row softmax, threshold -> adj + edge_attr -----
__global__ void __launch_bounds__(256)
k_scores(const _Float16* __restrict__ Xa, float* __restrict__ adj,
         float* __restrict__ eattr, const float* __restrict__ thr_p) {
    __shared__ float strip[16][400];
    __shared__ float rinv[16];
    int lane = threadIdx.x & 31, wave = threadIdx.x >> 5;
    int bt = blockIdx.x;
    const _Float16* Xbt = Xa + (long)bt * N_ * E_;
    float thr = 1.f / (1.f + __expf(-thr_p[0]));
    for (int it = 0; it < 25; ++it) {
        const _Float16* At = Xbt + it * 16 * E_;
        for (int j = wave; j < 25; j += 8) {
            const _Float16* Bt = Xbt + j * 16 * E_;
            v8f acc = {};
            acc = wmma32(load_a(At, E_, 0),  load_b(Bt, E_, 0),  acc);
            acc = wmma32(load_a(At, E_, 32), load_b(Bt, E_, 32), acc);
            int n = lane & 15, h = lane >> 4;
#pragma unroll
            for (int i = 0; i < 8; ++i)
                strip[i + 8 * h][j * 16 + n] = fmaxf(acc[i], 0.f);
        }
        __syncthreads();
        // stable row softmax: one 16-lane half-wave group per row
        {
            int g = threadIdx.x >> 4;   // row 0..15
            int id = threadIdx.x & 15;
            float mx = -1e30f;
            for (int n2 = id; n2 < 400; n2 += 16) mx = fmaxf(mx, strip[g][n2]);
#pragma unroll
            for (int off = 8; off; off >>= 1) mx = fmaxf(mx, __shfl_xor(mx, off, 16));
            float sum = 0.f;
            for (int n2 = id; n2 < 400; n2 += 16) {
                float e = __expf(strip[g][n2] - mx);
                strip[g][n2] = e;
                sum += e;
            }
#pragma unroll
            for (int off = 8; off; off >>= 1) sum += __shfl_xor(sum, off, 16);
            if (id == 0) rinv[g] = 1.f / sum;
        }
        __syncthreads();
        long base = (long)bt * (N_ * N_) + (long)it * 16 * N_;
        for (int idx = threadIdx.x; idx < 16 * 400; idx += 256) {
            int mrow = idx / 400, n2 = idx % 400;
            float v = fmaxf(strip[mrow][n2] * rinv[mrow] - thr, 0.f);
            adj[base + idx] = v;
            eattr[base + idx] = v;
        }
        __syncthreads();
    }
}

// ---- edge_index / batch (int32 patterns) -----------------------------------
__global__ void k_edges(int* __restrict__ ei, int* __restrict__ batch) {
    const long total = (long)BT_ * N_ * N_;
    for (long tid = (long)blockIdx.x * blockDim.x + threadIdx.x; tid < total;
         tid += (long)gridDim.x * blockDim.x) {
        int bt = (int)(tid / (N_ * N_));
        int rem = (int)(tid % (N_ * N_));
        int i = rem / N_, j = rem % N_;
        ei[tid] = bt * N_ + i;
        ei[total + tid] = bt * N_ + j;
        if (tid < (long)BT_ * N_) batch[tid] = (int)(tid / N_);
    }
}

// ---------------------------------------------------------------------------
// workspace layout (bytes, all 32B aligned)
static const size_t O_XSPH  = 0;                       // x_split f16    6,553,600
static const size_t O_WINH  = O_XSPH + 6553600;        // w_in f16          12,288
static const size_t O_WFC1H = O_WINH + 12288;          // w_fc1 f16         18,432
static const size_t O_WFC2H = O_WFC1H + 18432;         // w_fc2 f16         12,288
static const size_t O_CONVW = O_WFC2H + 12288;         // conv w f16       135,168
static const size_t O_H1    = O_CONVW + 135168;        // 9,830,400
static const size_t O_H2A   = O_H1 + 9830400;
static const size_t O_H2B   = O_H2A + 9830400;
static const size_t O_H3    = O_H2B + 9830400;
static const size_t O_XATT  = O_H3 + 9830400;          // 6,553,600
// d_out layout (float elements)
static const long OUT_NF    = 0;                       //  3,276,800
static const long OUT_ADJ   = 3276800;                 // 20,480,000
static const long OUT_EI    = 23756800;                // 40,960,000 (int32)
static const long OUT_EATTR = 64716800;                // 20,480,000
static const long OUT_BATCH = 85196800;                //     51,200

extern "C" void kernel_launch(void* const* d_in, const int* in_sizes, int n_in,
                              void* d_out, int out_size, void* d_ws, size_t ws_size,
                              hipStream_t stream) {
    // input order (setup_inputs insertion order, params flattened recursively)
    const float* x_split = (const float*)d_in[0];     // [8,400,16,64]
    const float* w_in    = (const float*)d_in[1];     // [96,64]
    const float* b_in    = (const float*)d_in[2];     // [96]
    const float* cw[6]   = {(const float*)d_in[3], (const float*)d_in[5],
                            (const float*)d_in[7], (const float*)d_in[9],
                            (const float*)d_in[11], (const float*)d_in[13]};
    const float* cb[6]   = {(const float*)d_in[4], (const float*)d_in[6],
                            (const float*)d_in[8], (const float*)d_in[10],
                            (const float*)d_in[12], (const float*)d_in[14]};
    const float* w_fc1   = (const float*)d_in[15];    // [96,96]
    const float* b_fc1   = (const float*)d_in[16];
    const float* w_fc2   = (const float*)d_in[17];    // [64,96]
    const float* b_fc2   = (const float*)d_in[18];
    const float* w_at1   = (const float*)d_in[19];    // [200,400]
    const float* w_at2   = (const float*)d_in[20];    // [400,200]
    const float* thresh  = (const float*)d_in[21];    // [1]

    char* ws = (char*)d_ws;
    _Float16* xsph  = (_Float16*)(ws + O_XSPH);
    _Float16* winh  = (_Float16*)(ws + O_WINH);
    _Float16* wfc1h = (_Float16*)(ws + O_WFC1H);
    _Float16* wfc2h = (_Float16*)(ws + O_WFC2H);
    _Float16* convw = (_Float16*)(ws + O_CONVW);
    _Float16* H1    = (_Float16*)(ws + O_H1);
    _Float16* H2a   = (_Float16*)(ws + O_H2A);
    _Float16* H2b   = (_Float16*)(ws + O_H2B);
    _Float16* H3    = (_Float16*)(ws + O_H3);
    _Float16* xatt  = (_Float16*)(ws + O_XATT);

    float* out   = (float*)d_out;
    float* nf    = out + OUT_NF;
    float* adj   = out + OUT_ADJ;
    int*   ei    = (int*)(out + OUT_EI);
    float* eattr = out + OUT_EATTR;
    int*   batch = (int*)(out + OUT_BATCH);

    // 1) f32 -> f16 staging
    k_cvt<<<12800, 256, 0, stream>>>(x_split, xsph, 3276800);
    k_cvt<<<24, 256, 0, stream>>>(w_in,  winh,  6144);
    k_cvt<<<36, 256, 0, stream>>>(w_fc1, wfc1h, 9216);
    k_cvt<<<24, 256, 0, stream>>>(w_fc2, wfc2h, 6144);
    // conv weights: [o][c][k] -> [tap][o][c], 11 tap blocks per layer
    const int taps[3] = {2, 3, 6}, tapoff[3] = {0, 2, 5};
    for (int l = 0; l < 2; ++l)
        for (int g = 0; g < 3; ++g)
            k_convw<<<72, 256, 0, stream>>>(cw[l * 3 + g],
                                            convw + l * 33792 + tapoff[g] * 3072,
                                            taps[g]);
    // 2) input linear: [51200,64] x [96,64]^T -> H1 f16 [51200,96]
    k_gemm<<<2400, 256, 0, stream>>>(xsph, winh, b_in, H1, nullptr,
                                     64, 96, /*relu=*/0, /*mode=*/0);
    // 3) MSTCN layers (dil 1, 2), WMMA per tap
    k_conv<<<400, 256, 0, stream>>>(H1,  H2a, convw,          cb[0], cb[1], cb[2], 1);
    k_conv<<<400, 256, 0, stream>>>(H2a, H2b, convw + 33792,  cb[3], cb[4], cb[5], 2);
    // 4) fc1 (relu) -> H3 f16; fc2 -> node_features f32 [B,T,N,E]
    k_gemm<<<2400, 256, 0, stream>>>(H2b, wfc1h, b_fc1, H3, nullptr, 96, 96, 1, 0);
    k_gemm<<<1600, 256, 0, stream>>>(H3,  wfc2h, b_fc2, nullptr, nf, 96, 64, 0, 1);
    // 5) spatial attention -> x_att f16
    k_attn<<<BT_, 256, 0, stream>>>(nf, w_at1, w_at2, xatt);
    // 6) relu(Xa Xa^T), softmax, threshold -> sparse_adj + edge_attr
    k_scores<<<BT_, 256, 0, stream>>>(xatt, adj, eattr, thresh);
    // 7) static edge_index / batch patterns
    k_edges<<<80000, 256, 0, stream>>>(ei, batch);
}
